// CrossSeqTransformer_20504173871396
// MI455X (gfx1250) — matrix operand order repaired and checked by reading, not compiled, not run on hardware
//
#include <hip/hip_runtime.h>
#include <hip/hip_bf16.h>
#include <math.h>

typedef __attribute__((ext_vector_type(16))) _Float16 v16h;
typedef __attribute__((ext_vector_type(8)))  float    v8f;

#define DEVFN static __device__ __forceinline__

// s_wait_asynccnt immediates for the 2-deep KV double buffer (4 async issues/tile)
#define ASYNC_WAIT_4() asm volatile("s_wait_asynccnt 0x4" ::: "memory")
#define ASYNC_WAIT_0() asm volatile("s_wait_asynccnt 0x0" ::: "memory")

DEVFN void async_b128_to_lds(unsigned lds_off, const float* gaddr) {
    // GLOBAL_LOAD_ASYNC_TO_LDS_B128: vdst = LDS byte offset, vaddr = 64-bit global addr
    asm volatile("global_load_async_to_lds_b128 %0, %1, off"
                 :: "v"(lds_off), "v"(gaddr) : "memory");
}

DEVFN unsigned lds_offset(const void* p) {
    // generic (flat) LDS pointer: addr[31:0] is the wave-relative LDS byte address
    return (unsigned)(unsigned long long)p;
}

DEVFN v8f wmma_f16(v16h a, v16h b, v8f c) {
    // D = A(16x32 f16) x B(32x16 f16) + C(16x16 f32)
    return __builtin_amdgcn_wmma_f32_16x16x32_f16(false, a, false, b, (short)0, c, false, false);
}

// ---- A-fragment (16x32 f16) from row-major f32 [rows][ld], row = lane%16.
// ISA layout: v<4 -> K = (hi?8:0)+2*(v&3); v>=4 -> K = 16+(hi?8:0)+2*(v&3)
DEVFN v16h frag_a_f32(const float* __restrict__ p, int ld, int row, int kk, bool hi, float scale) {
    v16h f;
    const float* rp = p + (size_t)row * ld;
#pragma unroll
    for (int v = 0; v < 8; ++v) {
        int k = kk + ((v & 4) ? 16 : 0) + (hi ? 8 : 0) + 2 * (v & 3);
        f[2 * v]     = (_Float16)(rp[k]     * scale);
        f[2 * v + 1] = (_Float16)(rp[k + 1] * scale);
    }
    return f;
}

// ---- B-fragment (32x16 f16) from transposed f16 weights [N][K], col = lane%16.
// ISA layout: v<4 -> K = (hi?8:0)+2*(v&3); v>=4 -> K = 16+(hi?8:0)+2*(v&3)
DEVFN v16h frag_b_f16t(const _Float16* __restrict__ p, int ld, int col, int kk, bool hi) {
    v16h f;
    const _Float16* rp = p + (size_t)col * ld;
#pragma unroll
    for (int v = 0; v < 8; ++v) {
        int k = kk + ((v & 4) ? 16 : 0) + (hi ? 8 : 0) + 2 * (v & 3);
        f[2 * v]     = rp[k];
        f[2 * v + 1] = rp[k + 1];
    }
    return f;
}

// =================== generic WMMA GEMM: C[M,N] = act(A[M,K] @ Bt^T + bias + resid) ====
// Bt is f16, stored transposed as [N][K]. One wave -> 16x64 tile (A frag reused 4x).
__global__ void gemm_wmma_kernel(const float* __restrict__ A, const _Float16* __restrict__ Bt,
                                 float* __restrict__ Cout, int M, int N, int K,
                                 const float* __restrict__ bias,
                                 const float* __restrict__ resid, int act_gelu) {
    const int m0   = blockIdx.x * 16;
    const int n0   = blockIdx.y * 64;
    const int lane = threadIdx.x & 31;
    const bool hi  = lane >= 16;
    const int l16  = lane & 15;
    const int arow = m0 + l16;

    v8f acc[4] = {};
    for (int kk = 0; kk < K; kk += 32) {
        if (kk + 32 < K)
            __builtin_prefetch(A + (size_t)arow * K + kk + 32, 0, 1);
        v16h a = frag_a_f32(A, K, arow, kk, hi, 1.0f);
#pragma unroll
        for (int t = 0; t < 4; ++t) {
            v16h b = frag_b_f16t(Bt, K, n0 + 16 * t + l16, kk, hi);
            acc[t] = wmma_f16(a, b, acc[t]);
        }
    }
#pragma unroll
    for (int t = 0; t < 4; ++t) {
        int col = n0 + 16 * t + l16;
        float bb = bias ? bias[col] : 0.0f;
#pragma unroll
        for (int i = 0; i < 8; ++i) {
            int rrow = m0 + (hi ? 8 : 0) + i;      // C/D layout: lanes<16 -> M=i, lanes>=16 -> M=8+i
            float x = acc[t][i] + bb;
            if (resid) x += resid[(size_t)rrow * N + col];
            if (act_gelu) x = 0.5f * x * (1.0f + erff(x * 0.70710678118654752f));
            Cout[(size_t)rrow * N + col] = x;
        }
    }
}

// =================== flash attention, one wave per (16-row Q block, head) ============
// K/V tiles are double-buffered in LDS via GLOBAL_LOAD_ASYNC_TO_LDS_B128 (ASYNCcnt),
// overlapping the next tile's fetch with this tile's softmax + 2 WMMAs.
__global__ void flash_attn_kernel(const float* __restrict__ Q, const float* __restrict__ Km,
                                  const float* __restrict__ Vm, const float* __restrict__ lr,
                                  const float* __restrict__ beta, int layer,
                                  float* __restrict__ O, int L, int C) {
    const int i0   = blockIdx.x * 16;
    const int h16  = blockIdx.y * 16;     // Dh = 16
    const int lane = threadIdx.x & 31;
    const bool hi  = lane >= 16;
    const int l16  = lane & 15;
    const float scale = 0.25f;            // Dh^-0.5

    __shared__ float    ldsK[2][16 * 16];
    __shared__ float    ldsV[2][16 * 16];
    __shared__ _Float16 pT[16 * 16];

    // issue one 16x16 f32 K-tile + V-tile into LDS buffer `buf` (4 async b128 issues)
    auto issue_kv = [&](int jb, int buf) {
        const int r = lane >> 2, seg = lane & 3;           // 4 lanes x 16B per row
        const size_t gofs = (size_t)(jb * 16 + r) * C + h16 + seg * 4;
        const unsigned lofs = (unsigned)(r * 64 + seg * 16);
        async_b128_to_lds(lds_offset(&ldsK[buf][0]) + lofs,       Km + gofs);
        async_b128_to_lds(lds_offset(&ldsV[buf][0]) + lofs,       Vm + gofs);
        async_b128_to_lds(lds_offset(&ldsK[buf][0]) + lofs + 512, Km + gofs + (size_t)8 * C);
        async_b128_to_lds(lds_offset(&ldsV[buf][0]) + lofs + 512, Vm + gofs + (size_t)8 * C);
    };

    // Q fragment (Dh=16 zero-padded to K=32), scale folded in
    v16h qf;
    {
        const float* rp = Q + (size_t)(i0 + l16) * C + h16;
#pragma unroll
        for (int v = 0; v < 8; ++v) {
            int k = ((v & 4) ? 16 : 0) + (hi ? 8 : 0) + 2 * (v & 3);
            qf[2 * v]     = (k     < 16) ? (_Float16)(rp[k]     * scale) : (_Float16)0.0f;
            qf[2 * v + 1] = (k + 1 < 16) ? (_Float16)(rp[k + 1] * scale) : (_Float16)0.0f;
        }
    }

    const float bta = beta[layer];
    float lrq[8], mrun[8], lsum[8];
#pragma unroll
    for (int i = 0; i < 8; ++i) {
        lrq[i]  = lr[i0 + (hi ? 8 : 0) + i];
        mrun[i] = -1e30f;
        lsum[i] = 0.0f;
    }
    v8f o = {};

    const int nkv = L / 16;
    issue_kv(0, 0);
    for (int jb = 0; jb < nkv; ++jb) {
        const int j0  = jb * 16;
        const int buf = jb & 1;
        if (jb + 1 < nkv) {
            issue_kv(jb + 1, buf ^ 1);   // prefetch next tile into the other buffer
            ASYNC_WAIT_4();              // oldest 4 (this tile) retired in-order
        } else {
            ASYNC_WAIT_0();
        }

        // K^T fragment: B[k=feature][n=key]; key row = l16, features contiguous in LDS
        v16h kf;
        {
            const float* kp = &ldsK[buf][l16 * 16];
#pragma unroll
            for (int v = 0; v < 8; ++v) {
                int k = ((v & 4) ? 16 : 0) + (hi ? 8 : 0) + 2 * (v & 3);
                kf[2 * v]     = (k     < 16) ? (_Float16)kp[k]     : (_Float16)0.0f;
                kf[2 * v + 1] = (k + 1 < 16) ? (_Float16)kp[k + 1] : (_Float16)0.0f;
            }
        }
        v8f s = {};
        s = wmma_f16(qf, kf, s);

        const float lrj = lr[j0 + l16];
        float p[8];
#pragma unroll
        for (int i = 0; i < 8; ++i) {
            float t = s[i] + bta * (lrq[i] + lrj);
            // row max across the 16 lanes of this half-wave (xor masks < 16 stay in-half)
            float rm = t;
            rm = fmaxf(rm, __shfl_xor(rm, 1, 32));
            rm = fmaxf(rm, __shfl_xor(rm, 2, 32));
            rm = fmaxf(rm, __shfl_xor(rm, 4, 32));
            rm = fmaxf(rm, __shfl_xor(rm, 8, 32));
            float mn  = fmaxf(mrun[i], rm);
            float fac = expf(mrun[i] - mn);
            mrun[i]   = mn;
            p[i]      = expf(t - mn);
            float rs  = p[i];
            rs += __shfl_xor(rs, 1, 32);
            rs += __shfl_xor(rs, 2, 32);
            rs += __shfl_xor(rs, 4, 32);
            rs += __shfl_xor(rs, 8, 32);
            lsum[i] = lsum[i] * fac + rs;
            o[i]   *= fac;
            pT[((hi ? 8 : 0) + i) * 16 + l16] = (_Float16)p[i];
        }
        __syncthreads();
        // P fragment (A layout, K=16 keys padded to 32)
        v16h pf;
#pragma unroll
        for (int v = 0; v < 8; ++v) {
            int k = ((v & 4) ? 16 : 0) + (hi ? 8 : 0) + 2 * (v & 3);
            pf[2 * v]     = (k     < 16) ? pT[l16 * 16 + k]     : (_Float16)0.0f;
            pf[2 * v + 1] = (k + 1 < 16) ? pT[l16 * 16 + k + 1] : (_Float16)0.0f;
        }
        // V fragment: B[k=key][n=dh] from LDS tile
        v16h vf;
#pragma unroll
        for (int v = 0; v < 8; ++v) {
            int k = ((v & 4) ? 16 : 0) + (hi ? 8 : 0) + 2 * (v & 3);
            vf[2 * v]     = (k     < 16) ? (_Float16)ldsV[buf][k * 16 + l16]       : (_Float16)0.0f;
            vf[2 * v + 1] = (k + 1 < 16) ? (_Float16)ldsV[buf][(k + 1) * 16 + l16] : (_Float16)0.0f;
        }
        o = wmma_f16(pf, vf, o);
        __syncthreads();
    }
#pragma unroll
    for (int i = 0; i < 8; ++i) {
        int rrow = i0 + (hi ? 8 : 0) + i;
        O[(size_t)rrow * C + h16 + l16] = o[i] / lsum[i];
    }
}

// =================== LayerNorm over C=128, one wave per row =========================
__global__ void ln_kernel(const float* __restrict__ x, const float* __restrict__ g,
                          const float* __restrict__ b, float* __restrict__ out) {
    const int row  = blockIdx.x;
    const int lane = threadIdx.x & 31;
    const float* xp = x + (size_t)row * 128;
    float v[4];
    float s = 0.0f, ss = 0.0f;
#pragma unroll
    for (int j = 0; j < 4; ++j) {
        v[j] = xp[lane + 32 * j];
        s += v[j];
        ss += v[j] * v[j];
    }
#pragma unroll
    for (int m = 1; m < 32; m <<= 1) {
        s  += __shfl_xor(s, m, 32);
        ss += __shfl_xor(ss, m, 32);
    }
    float mean = s * (1.0f / 128.0f);
    float var  = ss * (1.0f / 128.0f) - mean * mean;   // biased, matches jnp.var
    float inv  = rsqrtf(var + 1e-5f);
    float* op = out + (size_t)row * 128;
#pragma unroll
    for (int j = 0; j < 4; ++j) {
        int c = lane + 32 * j;
        op[c] = (v[j] - mean) * inv * g[c] + b[c];
    }
}

// =================== data-movement kernels ==========================================
__global__ void tokenize_kernel(const float* __restrict__ z, const float* __restrict__ se,
                                float* __restrict__ tok, int S, int C, int Ntok) {
    int idx = blockIdx.x * blockDim.x + threadIdx.x;
    int total = S * Ntok * C;
    if (idx >= total) return;
    int l = idx / C, c = idx % C;
    int s = l / Ntok, n = l % Ntok;
    tok[idx] = z[((size_t)s * C + c) * Ntok + n] + se[s * C + c];
}

__global__ void logr_kernel(const float* __restrict__ r, float* __restrict__ lr,
                            int Ntok, int Ltot) {
    int idx = blockIdx.x * blockDim.x + threadIdx.x;
    if (idx >= Ltot) return;
    lr[idx] = logf(fmaxf(r[idx / Ntok], 1e-8f));
}

__global__ void detok_kernel(const float* __restrict__ tok, float* __restrict__ out,
                             int S, int C, int Ntok) {
    int idx = blockIdx.x * blockDim.x + threadIdx.x;   // idx = (s*C + c)*Ntok + n
    int total = S * C * Ntok;
    if (idx >= total) return;
    int s = idx / (C * Ntok);
    int rem = idx % (C * Ntok);
    int c = rem / Ntok, n = rem % Ntok;
    out[idx] = tok[((size_t)s * Ntok + n) * C + c];
}

// transpose-convert f32 [K][N] -> f16 [N][K]
__global__ void convt_kernel(const float* __restrict__ in, _Float16* __restrict__ out,
                             int K, int N) {
    int idx = blockIdx.x * blockDim.x + threadIdx.x;
    if (idx >= K * N) return;
    int k = idx / N, n = idx % N;
    out[(size_t)n * K + k] = (_Float16)in[idx];
}

// =================== host orchestration ============================================
extern "C" void kernel_launch(void* const* d_in, const int* in_sizes, int n_in,
                              void* d_out, int out_size, void* d_ws, size_t ws_size,
                              hipStream_t stream) {
    (void)in_sizes; (void)n_in; (void)out_size; (void)ws_size;
    const int S = 4, C = 128, Ntok = 1000, L = S * Ntok, nL = 2, H = 8, FF = 4 * C;

    const float* z      = (const float*)d_in[0];
    const float* r      = (const float*)d_in[1];
    const float* se     = (const float*)d_in[2];
    const float* Wq     = (const float*)d_in[3];
    const float* Wk     = (const float*)d_in[4];
    const float* Wv     = (const float*)d_in[5];
    const float* Wp     = (const float*)d_in[6];
    const float* beta   = (const float*)d_in[7];
    const float* ln1_g  = (const float*)d_in[8];
    const float* ln1_b  = (const float*)d_in[9];
    const float* ffn_w1 = (const float*)d_in[10];
    const float* ffn_b1 = (const float*)d_in[11];
    const float* ffn_w2 = (const float*)d_in[12];
    const float* ffn_b2 = (const float*)d_in[13];
    const float* ln2_g  = (const float*)d_in[14];
    const float* ln2_b  = (const float*)d_in[15];
    float* out = (float*)d_out;

    // workspace carve-out (256B aligned)
    char* base = (char*)d_ws;
    size_t off = 0;
    auto carve = [&](size_t bytes) -> char* {
        char* p = base + off;
        off = (off + bytes + 255) & ~((size_t)255);
        return p;
    };
    float*     tokens = (float*)carve((size_t)L * C * 4);
    float*     lr     = (float*)carve((size_t)L * 4);
    float*     tmp    = (float*)carve((size_t)L * C * 4);
    float*     Qb     = (float*)carve((size_t)L * C * 4);
    float*     Kb     = (float*)carve((size_t)L * C * 4);
    float*     Vb     = (float*)carve((size_t)L * C * 4);
    float*     attO   = (float*)carve((size_t)L * C * 4);
    float*     Hbuf   = (float*)carve((size_t)L * FF * 4);
    _Float16*  WqT    = (_Float16*)carve((size_t)nL * C * C * 2);
    _Float16*  WkT    = (_Float16*)carve((size_t)nL * C * C * 2);
    _Float16*  WvT    = (_Float16*)carve((size_t)nL * C * C * 2);
    _Float16*  WpT    = (_Float16*)carve((size_t)nL * C * C * 2);
    _Float16*  w1T    = (_Float16*)carve((size_t)C * FF * 2);
    _Float16*  w2T    = (_Float16*)carve((size_t)FF * C * 2);

    const int TPB = 256;
    // tokenize + log-reliability
    tokenize_kernel<<<(L * C + TPB - 1) / TPB, TPB, 0, stream>>>(z, se, tokens, S, C, Ntok);
    logr_kernel<<<(L + TPB - 1) / TPB, TPB, 0, stream>>>(r, lr, Ntok, L);
    // convert weights to transposed f16
    for (int l = 0; l < nL; ++l) {
        size_t o = (size_t)l * C * C;
        convt_kernel<<<(C * C + TPB - 1) / TPB, TPB, 0, stream>>>(Wq + o, WqT + o, C, C);
        convt_kernel<<<(C * C + TPB - 1) / TPB, TPB, 0, stream>>>(Wk + o, WkT + o, C, C);
        convt_kernel<<<(C * C + TPB - 1) / TPB, TPB, 0, stream>>>(Wv + o, WvT + o, C, C);
        convt_kernel<<<(C * C + TPB - 1) / TPB, TPB, 0, stream>>>(Wp + o, WpT + o, C, C);
    }
    convt_kernel<<<(C * FF + TPB - 1) / TPB, TPB, 0, stream>>>(ffn_w1, w1T, C, FF);
    convt_kernel<<<(FF * C + TPB - 1) / TPB, TPB, 0, stream>>>(ffn_w2, w2T, FF, C);

    const dim3 gC(L / 16, C / 64);    // N = 128
    const dim3 gF(L / 16, FF / 64);   // N = 512
    for (int l = 0; l < nL; ++l) {
        size_t o = (size_t)l * C * C;
        gemm_wmma_kernel<<<gC, 32, 0, stream>>>(tokens, WqT + o, Qb, L, C, C, nullptr, nullptr, 0);
        gemm_wmma_kernel<<<gC, 32, 0, stream>>>(tokens, WkT + o, Kb, L, C, C, nullptr, nullptr, 0);
        gemm_wmma_kernel<<<gC, 32, 0, stream>>>(tokens, WvT + o, Vb, L, C, C, nullptr, nullptr, 0);
        flash_attn_kernel<<<dim3(L / 16, H), 32, 0, stream>>>(Qb, Kb, Vb, lr, beta, l, attO, L, C);
        gemm_wmma_kernel<<<gC, 32, 0, stream>>>(attO, WpT + o, tmp, L, C, C, nullptr, tokens, 0);
        ln_kernel<<<L, 32, 0, stream>>>(tmp, ln1_g + (size_t)l * C, ln1_b + (size_t)l * C, tokens);
    }
    // FFN + LN
    gemm_wmma_kernel<<<gF, 32, 0, stream>>>(tokens, w1T, Hbuf, L, FF, C, ffn_b1, nullptr, 1);
    gemm_wmma_kernel<<<gC, 32, 0, stream>>>(Hbuf, w2T, tmp, L, C, FF, ffn_b2, tokens, 0);
    ln_kernel<<<L, 32, 0, stream>>>(tmp, ln2_g, ln2_b, tokens);
    // de-tokenize to output
    detok_kernel<<<(S * C * Ntok + TPB - 1) / TPB, TPB, 0, stream>>>(tokens, out, S, C, Ntok);
}
